// PatternAwareMixtureOfExperts_83846351553182
// MI455X (gfx1250) — matrix operand order repaired and checked by reading, not compiled, not run on hardware
//
#include <hip/hip_runtime.h>

// ---------------- problem constants ----------------
#define Bb   8
#define Ss   4096
#define Dd   512
#define Hh   2048
#define Ee   8
#define Pp   16
#define Tt   (Bb*Ss)        // 32768 tokens
#define TD   ((size_t)Tt*Dd)

typedef __attribute__((ext_vector_type(16))) __bf16 v16bf;
typedef __attribute__((ext_vector_type(8)))  float  v8f;
typedef __attribute__((ext_vector_type(4)))  int    v4i;

__device__ __forceinline__ unsigned short f2bf(float f) {
    unsigned int u = __float_as_uint(f);
    unsigned int r = (u + 0x7FFFu + ((u >> 16) & 1u)) >> 16;  // RNE
    return (unsigned short)r;
}

__device__ __forceinline__ v8f wmma_bf16(v16bf a, v16bf b, v8f c) {
    return __builtin_amdgcn_wmma_f32_16x16x32_bf16(false, a, false, b, (short)0, c, false, false);
}

// ---- CDNA5 async global->LDS copy (ASYNCcnt path), with safe fallback ----
#if defined(__has_builtin)
#if __has_builtin(__builtin_amdgcn_global_load_async_to_lds_b128) && \
    __has_builtin(__builtin_amdgcn_s_wait_asynccnt)
#define HAVE_ASYNC_LDS 1
#endif
#endif

__device__ __forceinline__ void cp16_g2l(unsigned short* lds_dst, const unsigned short* gsrc) {
#ifdef HAVE_ASYNC_LDS
    __builtin_amdgcn_global_load_async_to_lds_b128(
        (__attribute__((address_space(1))) v4i*)gsrc,
        (__attribute__((address_space(3))) v4i*)lds_dst, 0, 0);
#else
    *(uint4*)lds_dst = *(const uint4*)gsrc;
#endif
}

__device__ __forceinline__ void cp_wait_async() {
#ifdef HAVE_ASYNC_LDS
    __builtin_amdgcn_s_wait_asynccnt(0);
#endif
}

// ---------------- workspace layout (bytes) ----------------
#define WS_COUNTS   0
#define WS_ZSUM     64
#define WS_BAL      128
#define WS_TOK      256
#define WS_GATE     (WS_TOK  + (size_t)Ee*Tt*4)
#define WS_XBF      (WS_GATE + (size_t)Ee*Tt*4)
#define WS_W1P      (WS_XBF  + (size_t)Tt*Dd*2)
#define WS_W3P      (WS_W1P  + (size_t)Ee*Dd*Hh*2)
#define WS_W2P      (WS_W3P  + (size_t)Ee*Dd*Hh*2)

// ---------------- init: zero y + accumulators ----------------
__global__ void k_init(float* __restrict__ y, int* __restrict__ counts,
                       float* __restrict__ zsum, float* __restrict__ bal) {
    size_t i = (size_t)blockIdx.x * blockDim.x + threadIdx.x;
    size_t stride = (size_t)gridDim.x * blockDim.x;
    for (size_t k = i; k < TD; k += stride) y[k] = 0.0f;
    if (blockIdx.x == 0) {
        if (threadIdx.x < Ee) { counts[threadIdx.x] = 0; bal[threadIdx.x] = 0.0f; }
        if (threadIdx.x == Ee) zsum[0] = 0.0f;
    }
}

// ---------------- convert x -> bf16 ----------------
__global__ void k_cvt_x(const float* __restrict__ x, unsigned short* __restrict__ xbf) {
    size_t i = (size_t)blockIdx.x * blockDim.x + threadIdx.x;
    size_t stride = (size_t)gridDim.x * blockDim.x;
    for (size_t k = i; k < TD; k += stride) xbf[k] = f2bf(x[k]);
}

// ---------------- pack weight [E,K,N] into WMMA B-fragment layout ----------------
__global__ void k_pack_w(const float* __restrict__ src, unsigned short* __restrict__ dst,
                         int K, int N) {
    int kT = K / 32, nT = N / 16;
    size_t total = (size_t)Ee * nT * kT * 32;
    size_t i = (size_t)blockIdx.x * blockDim.x + threadIdx.x;
    size_t stride = (size_t)gridDim.x * blockDim.x;
    for (size_t idx = i; idx < total; idx += stride) {
        int L  = (int)(idx & 31);
        size_t r = idx >> 5;
        int kt = (int)(r % kT); r /= kT;
        int nt = (int)(r % nT);
        int ex = (int)(r / nT);
        int n     = nt * 16 + (L & 15);
        int kbase = kt * 32 + ((L >> 4) << 3);
        const float* se = src + (size_t)ex * K * N;
        unsigned short* de = dst + idx * 16;
#pragma unroll
        for (int q = 0; q < 16; ++q)
            de[q] = f2bf(se[(size_t)(kbase + q + ((q >> 3) << 3)) * N + n]);
    }
}

// ---------------- router ----------------
__global__ __launch_bounds__(256) void k_router(
    const float* __restrict__ x, const int* __restrict__ pids,
    const float* __restrict__ Wr, const float* __restrict__ Wp,
    int* __restrict__ counts, int* __restrict__ tokList, float* __restrict__ gateVal,
    float* __restrict__ zsum, float* __restrict__ bal) {
    int t = blockIdx.x * 256 + threadIdx.x;
    float lg[8];
#pragma unroll
    for (int e = 0; e < 8; ++e) lg[e] = 0.0f;
    const float* xr = x + (size_t)t * Dd;
    for (int d = 0; d < Dd; ++d) {
        float xv = xr[d];
        const float* wr = Wr + d * Ee;
#pragma unroll
        for (int e = 0; e < 8; ++e) lg[e] += xv * wr[e];
    }
    int pid = pids[t / Ss];
#pragma unroll
    for (int e = 0; e < 8; ++e) lg[e] += Wp[pid * Ee + e];

    float zs = 0.0f;
#pragma unroll
    for (int e = 0; e < 8; ++e) zs += lg[e] * lg[e];

    float mx = lg[0];
#pragma unroll
    for (int e = 1; e < 8; ++e) mx = fmaxf(mx, lg[e]);
    float pe[8], se = 0.0f;
#pragma unroll
    for (int e = 0; e < 8; ++e) { pe[e] = __expf(lg[e] - mx); se += pe[e]; }
    float inv = 1.0f / se;
#pragma unroll
    for (int e = 0; e < 8; ++e) pe[e] *= inv;

    int i0 = 0; float v0 = lg[0];
#pragma unroll
    for (int e = 1; e < 8; ++e) if (lg[e] > v0) { v0 = lg[e]; i0 = e; }
    int i1 = -1; float v1 = -3.4e38f;
#pragma unroll
    for (int e = 0; e < 8; ++e) if (e != i0 && lg[e] > v1) { v1 = lg[e]; i1 = e; }
    float g0 = 1.0f / (1.0f + __expf(v1 - v0));
    float g1 = 1.0f - g0;

    int p0 = atomicAdd(&counts[i0], 1);
    tokList[(size_t)i0 * Tt + p0] = t;  gateVal[(size_t)i0 * Tt + p0] = g0;
    int p1 = atomicAdd(&counts[i1], 1);
    tokList[(size_t)i1 * Tt + p1] = t;  gateVal[(size_t)i1 * Tt + p1] = g1;

    __shared__ float sZ;
    __shared__ float sB[8];
    if (threadIdx.x == 0) sZ = 0.0f;
    if (threadIdx.x < 8) sB[threadIdx.x] = 0.0f;
    __syncthreads();
    atomicAdd(&sZ, zs);
#pragma unroll
    for (int e = 0; e < 8; ++e) atomicAdd(&sB[e], pe[e]);
    __syncthreads();
    if (threadIdx.x == 0) atomicAdd(zsum, sZ);
    if (threadIdx.x < 8) atomicAdd(&bal[threadIdx.x], sB[threadIdx.x]);
}

// ---------------- aux losses ----------------
__global__ void k_losses(const float* __restrict__ Wp, const float* __restrict__ zsum,
                         const float* __restrict__ bal, float* __restrict__ outTail) {
    if (threadIdx.x != 0 || blockIdx.x != 0) return;
    float z = zsum[0] / (float)((size_t)Tt * Ee) * 0.001f;
    float bl = 0.0f;
    for (int e = 0; e < Ee; ++e) {
        float m = bal[e] / (float)Tt - 1.0f / (float)Ee;
        bl += m * m;
    }
    bl /= (float)Ee;
    float Pn[Pp][Ee];
    for (int p = 0; p < Pp; ++p) {
        float mx = Wp[p * Ee];
        for (int e = 1; e < Ee; ++e) mx = fmaxf(mx, Wp[p * Ee + e]);
        float s = 0.0f;
        for (int e = 0; e < Ee; ++e) { Pn[p][e] = __expf(Wp[p * Ee + e] - mx); s += Pn[p][e]; }
        float inv = 1.0f / s;
        for (int e = 0; e < Ee; ++e) Pn[p][e] *= inv;
    }
    float s = 0.0f;
    for (int p = 0; p < Pp; ++p)
        for (int q = 0; q < Pp; ++q) {
            if (p == q) continue;
            float d = 0.0f;
            for (int e = 0; e < Ee; ++e) d += Pn[p][e] * Pn[q][e];
            s += d;
        }
    float spec = s / (float)(Pp * Pp) * 0.1f;
    outTail[0] = z; outTail[1] = bl; outTail[2] = spec;
}

// ---------------- fused SwiGLU expert FFN, M=32 tokens/block ----------------
// LDS: sA 32KB (2 mSub x 16 kt x 32 lanes x 16 bf16), sH 16KB (2 mSub x 8 kt2 x 32 x 16)
#define MTOK 32
#define MAXTILES (Tt / MTOK)   // 1024 per expert
#define NCHUNK 8               // H split into 8 chunks of 256 columns

__global__ __launch_bounds__(256) void k_ffn(
    const unsigned short* __restrict__ xbf,
    const unsigned short* __restrict__ w1p,
    const unsigned short* __restrict__ w3p,
    const unsigned short* __restrict__ w2p,
    const int* __restrict__ counts, const int* __restrict__ tokList,
    const float* __restrict__ gateVal, float* __restrict__ y) {
    int ex   = blockIdx.x / MAXTILES;
    int tile = blockIdx.x % MAXTILES;
    int cnt  = counts[ex];
    if (tile * MTOK >= cnt) return;

    __shared__ __align__(32) unsigned short sA[2 * 16 * 32 * 16];  // 32KB
    __shared__ __align__(32) unsigned short sH[2 * 8 * 32 * 16];   // 16KB
    __shared__ int   sTok[MTOK];
    __shared__ float sG[MTOK];

    int tid  = threadIdx.x;
    int lane = tid & 31;
    int wv   = tid >> 5;   // 8 waves

    if (tid < MTOK) {
        int idx = tile * MTOK + tid;
        if (idx < cnt) {
            sTok[tid] = tokList[(size_t)ex * Tt + idx];
            sG[tid]   = gateVal[(size_t)ex * Tt + idx];
        } else { sTok[tid] = 0; sG[tid] = 0.0f; }
    }
    __syncthreads();

    // pack 32 token rows (D=512) into A-fragment layout: [ms][kt][L][i]
    // async global->LDS (ASYNCcnt) when available, else plain ld/st
    for (int e2 = tid; e2 < 2 * 16 * 32; e2 += 256) {
        int L  = e2 & 31;
        int kt = (e2 >> 5) & 15;
        int ms = e2 >> 9;
        int tok = sTok[ms * 16 + (L & 15)];
        int kbase = kt * 32 + ((L >> 4) << 3);   // 0 or 8
        const unsigned short* src = xbf + (size_t)tok * Dd;
        unsigned short* dst = sA + (size_t)e2 * 16;
        cp16_g2l(dst,      src + kbase);        // k = kbase .. +7
        cp16_g2l(dst + 8,  src + kbase + 16);   // k = kbase+16 .. +23
    }
    cp_wait_async();
    __syncthreads();

    // phase-2 accumulators: wave owns D n-tiles wv*4..wv*4+3, BOTH mSubs
    // (each W2 B-fragment loaded once, feeds 2 WMMAs)
    v8f yc0[4], yc1[4];
#pragma unroll
    for (int j = 0; j < 4; ++j) { yc0[j] = (v8f)(0.0f); yc1[j] = (v8f)(0.0f); }

    int mLane = (lane >> 4) * 8;     // row base within 16-token subtile (C layout)

    for (int ch = 0; ch < NCHUNK; ++ch) {
        // ---- phase 1: h[32, 256] for this chunk ----
        // wave wv covers local n-tiles {2wv, 2wv+1} x both mSubs (B-frag register reuse)
#pragma unroll
        for (int jn = 0; jn < 2; ++jn) {
            int ntl = wv * 2 + jn;          // 0..15 within chunk
            int ntg = ch * 16 + ntl;        // global over H/16 = 128
            v8f a1m0 = (v8f)(0.0f), a3m0 = (v8f)(0.0f);
            v8f a1m1 = (v8f)(0.0f), a3m1 = (v8f)(0.0f);
#pragma unroll 4
            for (int kt = 0; kt < 16; ++kt) {
                size_t wofs = ((((size_t)ex * 128 + ntg) * 16 + kt) * 32 + lane) * 16;
                const unsigned short* p1 = w1p + wofs;
                const unsigned short* p3 = w3p + wofs;
                __builtin_prefetch(p1 + 512, 0, 1);
                v16bf b1 = *(const v16bf*)p1;
                v16bf b3 = *(const v16bf*)p3;
                v16bf af0 = *(const v16bf*)(sA + (((size_t)0 * 16 + kt) * 32 + lane) * 16);
                v16bf af1 = *(const v16bf*)(sA + (((size_t)1 * 16 + kt) * 32 + lane) * 16);
                a1m0 = wmma_bf16(af0, b1, a1m0);
                a3m0 = wmma_bf16(af0, b3, a3m0);
                a1m1 = wmma_bf16(af1, b1, a1m1);
                a3m1 = wmma_bf16(af1, b3, a3m1);
            }
            // fuse SwiGLU, store both mSubs to sH in phase-2 A-fragment layout
            int cl   = ntl * 16 + (lane & 15);  // column within chunk (0..255)
            int kt2l = cl >> 5;                 // 0..7
            int rem  = cl & 31;
            int half, ii;
            if      (rem < 8)  { half = 0; ii = rem; }
            else if (rem < 16) { half = 1; ii = rem - 8; }
            else if (rem < 24) { half = 0; ii = rem - 8; }
            else               { half = 1; ii = rem - 16; }
#pragma unroll
            for (int vr = 0; vr < 8; ++vr) {
                int L2 = (mLane + vr) + half * 16;
                float v3 = a3m0[vr];
                float f0 = a1m0[vr] * (v3 / (1.0f + __expf(-v3)));
                sH[(((size_t)0 * 8 + kt2l) * 32 + L2) * 16 + ii] = f2bf(f0);
                float w3v = a3m1[vr];
                float f1 = a1m1[vr] * (w3v / (1.0f + __expf(-w3v)));
                sH[(((size_t)1 * 8 + kt2l) * 32 + L2) * 16 + ii] = f2bf(f1);
            }
        }
        __syncthreads();
        // ---- phase 2: y += h_chunk @ W2_chunk ; W2 frag reused across both mSubs ----
#pragma unroll
        for (int jt = 0; jt < 4; ++jt) {
            int nt = wv * 4 + jt;            // 0..31 over D/16
            v8f c0 = yc0[jt], c1 = yc1[jt];
#pragma unroll 4
            for (int kt2 = 0; kt2 < 8; ++kt2) {
                int ktg = ch * 8 + kt2;
                v16bf bw = *(const v16bf*)(
                    w2p + ((((size_t)ex * 32 + nt) * 64 + ktg) * 32 + lane) * 16);
                v16bf ah0 = *(const v16bf*)(sH + (((size_t)0 * 8 + kt2) * 32 + lane) * 16);
                v16bf ah1 = *(const v16bf*)(sH + (((size_t)1 * 8 + kt2) * 32 + lane) * 16);
                c0 = wmma_bf16(ah0, bw, c0);
                c1 = wmma_bf16(ah1, bw, c1);
            }
            yc0[jt] = c0; yc1[jt] = c1;
        }
        __syncthreads();   // before next chunk overwrites sH
    }

    // ---- gate + scatter-add to y ----
    int ncol = lane & 15;
#pragma unroll
    for (int jt = 0; jt < 4; ++jt) {
        int col = (wv * 4 + jt) * 16 + ncol;
#pragma unroll
        for (int vr = 0; vr < 8; ++vr) {
            int m0 = mLane + vr;             // mSub 0
            int m1 = 16 + mLane + vr;        // mSub 1
            atomicAdd(&y[(size_t)sTok[m0] * Dd + col], sG[m0] * yc0[jt][vr]);
            atomicAdd(&y[(size_t)sTok[m1] * Dd + col], sG[m1] * yc1[jt][vr]);
        }
    }
}

// ---------------- launcher ----------------
extern "C" void kernel_launch(void* const* d_in, const int* in_sizes, int n_in,
                              void* d_out, int out_size, void* d_ws, size_t ws_size,
                              hipStream_t stream) {
    const float* x   = (const float*)d_in[0];
    const int*   pid = (const int*)  d_in[1];
    const float* Wr  = (const float*)d_in[2];
    const float* Wp  = (const float*)d_in[3];
    const float* W1  = (const float*)d_in[4];
    const float* W2  = (const float*)d_in[5];
    const float* W3  = (const float*)d_in[6];

    char* ws = (char*)d_ws;
    int*   counts  = (int*)  (ws + WS_COUNTS);
    float* zsum    = (float*)(ws + WS_ZSUM);
    float* bal     = (float*)(ws + WS_BAL);
    int*   tokList = (int*)  (ws + WS_TOK);
    float* gateVal = (float*)(ws + WS_GATE);
    unsigned short* xbf = (unsigned short*)(ws + WS_XBF);
    unsigned short* w1p = (unsigned short*)(ws + WS_W1P);
    unsigned short* w3p = (unsigned short*)(ws + WS_W3P);
    unsigned short* w2p = (unsigned short*)(ws + WS_W2P);

    float* y       = (float*)d_out;
    float* outTail = y + TD;

    k_init <<<4096, 256, 0, stream>>>(y, counts, zsum, bal);
    k_cvt_x<<<4096, 256, 0, stream>>>(x, xbf);
    k_pack_w<<<2048, 256, 0, stream>>>(W1, w1p, Dd, Hh);
    k_pack_w<<<2048, 256, 0, stream>>>(W3, w3p, Dd, Hh);
    k_pack_w<<<2048, 256, 0, stream>>>(W2, w2p, Hh, Dd);
    k_router<<<Tt / 256, 256, 0, stream>>>(x, pid, Wr, Wp, counts, tokList, gateVal, zsum, bal);
    k_losses<<<1, 32, 0, stream>>>(Wp, zsum, bal, outTail);
    k_ffn   <<<Ee * MAXTILES, 256, 0, stream>>>(xbf, w1p, w3p, w2p, counts, tokList, gateVal, y);
}
